// CWRRTEWindowCell_12283606467875
// MI455X (gfx1250) — matrix operand order repaired and checked by reading, not compiled, not run on hardware
//
#include <hip/hip_runtime.h>
#include <hip/hip_bf16.h>
#include <math.h>

// ---------------------------------------------------------------------------
// CDNA5 (gfx1250) wave32 WMMA types
// ---------------------------------------------------------------------------
typedef __attribute__((ext_vector_type(16))) __bf16 v16bf;
typedef __attribute__((ext_vector_type(8)))  float  v8f;

union FragB { v16bf v; uint4 u[2]; };
union Acc   { v8f   v; float f[8]; };

static __device__ inline v8f wmma_bf16(v16bf a, v16bf b, v8f c) {
    // D = A(16x32) * B(32x16) + C, bf16 in, f32 accumulate
    return __builtin_amdgcn_wmma_f32_16x16x32_bf16(
        /*neg_a=*/false, a, /*neg_b=*/false, b,
        /*c_mod=*/(short)0, c, /*reuse_a=*/false, /*reuse_b=*/false);
}

static __device__ inline unsigned short f2bf(float f) {
    unsigned int u = __float_as_uint(f);
    u += 0x7FFFu + ((u >> 16) & 1u);         // round-to-nearest-even
    return (unsigned short)(u >> 16);
}
static __device__ inline float bf2f(unsigned short h) {
    return __uint_as_float(((unsigned int)h) << 16);
}

// Raw 32-bit LDS byte offset of a pointer to __shared__ memory
// (generic -> addrspace(3) cast, then ptrtoint).
static __device__ inline unsigned lds_off(const void* p) {
    return (unsigned)(unsigned long long)(__attribute__((address_space(3))) const void*)p;
}

// CDNA5 async global -> LDS copy, 16B per lane, tracked by ASYNCcnt.
static __device__ inline void async_copy_b128(unsigned lds_addr, const void* gptr) {
    asm volatile("global_load_async_to_lds_b128 %0, %1, off"
                 :: "v"(lds_addr), "v"(gptr)
                 : "memory");
}
static __device__ inline void wait_asynccnt0() {
    asm volatile("s_wait_asynccnt 0x0" ::: "memory");
}

// Problem constants
#define BB 32
#define WW 512
#define DD 512
#define HH 8
#define KK 64
#define DH 64
#define MM (BB * WW)   // 16384 rows

// ---------------------------------------------------------------------------
// Weight transpose + f32 -> bf16  (W: Kd x Nd row-major -> Wt: Nd x Kd)
// ---------------------------------------------------------------------------
__global__ void k_transpose_bf16(const float* __restrict__ Wm,
                                 unsigned short* __restrict__ Wt,
                                 int Kd, int Nd) {
    int idx = blockIdx.x * blockDim.x + threadIdx.x;
    int total = Kd * Nd;
    if (idx >= total) return;
    int k = idx / Nd, n = idx - k * Nd;
    Wt[(size_t)n * Kd + k] = f2bf(Wm[idx]);
}

// ---------------------------------------------------------------------------
// RMS norm rows of f32 (row length 512) -> bf16
// ---------------------------------------------------------------------------
__global__ __launch_bounds__(256)
void k_rmsnorm_bf16(const float* __restrict__ in, const float* __restrict__ scale,
                    unsigned short* __restrict__ out) {
    int row = blockIdx.x, tid = threadIdx.x;
    const float* x = in + (size_t)row * DD;
    float ss = 0.f;
    for (int d = tid; d < DD; d += 256) { float v = x[d]; ss += v * v; }
    __shared__ float red[256];
    red[tid] = ss; __syncthreads();
    for (int s = 128; s > 0; s >>= 1) { if (tid < s) red[tid] += red[tid + s]; __syncthreads(); }
    float inv = rsqrtf(red[0] / (float)DD + 1e-6f);
    for (int d = tid; d < DD; d += 256)
        out[(size_t)row * DD + d] = f2bf(x[d] * inv * scale[d]);
}

// ---------------------------------------------------------------------------
// Generic WMMA GEMM:  C(MxN) = A(MxKd, bf16) * Bt(NxKd, bf16)^T  + epilogue
// Block: 256 thr = 8 waves, tile 128x64, wave tile 32x32, K-step 32.
// Double-buffered LDS, staged with async global->LDS DMA (ASYNCcnt).
// EPI 0: f32 = acc+bias   EPI 1: f32 = acc+bias+res
// EPI 2: bf16 = acc+bias  EPI 3: bf16 = silu(acc1)*acc2   (dual B)
// ---------------------------------------------------------------------------
template <int EPI>
__global__ __launch_bounds__(256)
void k_gemm(const unsigned short* __restrict__ A,
            const unsigned short* __restrict__ Bt,
            const unsigned short* __restrict__ Bt2,
            const float* __restrict__ bias,
            const float* __restrict__ res,
            float* __restrict__ Cf,
            unsigned short* __restrict__ Cb,
            int M, int N, int Kd) {
    __shared__ __align__(16) unsigned short sA[2][128 * 32];
    __shared__ __align__(16) unsigned short sB[2][64 * 32];
    __shared__ __align__(16) unsigned short sB2[(EPI == 3) ? 2 : 1][(EPI == 3) ? 64 * 32 : 16];

    int tid = threadIdx.x;
    int lane = tid & 31, wave = tid >> 5;
    int half = lane >> 4, lx = lane & 15;
    int waveM = wave >> 1, waveN = wave & 1;
    int m0 = blockIdx.y * 128, n0 = blockIdx.x * 64;

    unsigned aBase  = lds_off(&sA[0][0]);
    unsigned bBase  = lds_off(&sB[0][0]);
    unsigned b2Base = lds_off(&sB2[0][0]);

    // per-thread staging coordinates
    int cA0 = tid * 2;
    int rowA0 = cA0 >> 2, segA0 = cA0 & 3;
    int rowA1 = (cA0 + 1) >> 2, segA1 = (cA0 + 1) & 3;
    int rowB = tid >> 2, segB = tid & 3;

    auto stage = [&](int buf, int kk) {
        async_copy_b128(aBase + (unsigned)(buf * 128 * 32 + rowA0 * 32 + segA0 * 8) * 2u,
                        &A[(size_t)(m0 + rowA0) * Kd + kk + segA0 * 8]);
        async_copy_b128(aBase + (unsigned)(buf * 128 * 32 + rowA1 * 32 + segA1 * 8) * 2u,
                        &A[(size_t)(m0 + rowA1) * Kd + kk + segA1 * 8]);
        async_copy_b128(bBase + (unsigned)(buf * 64 * 32 + rowB * 32 + segB * 8) * 2u,
                        &Bt[(size_t)(n0 + rowB) * Kd + kk + segB * 8]);
        if constexpr (EPI == 3)
            async_copy_b128(b2Base + (unsigned)(buf * 64 * 32 + rowB * 32 + segB * 8) * 2u,
                            &Bt2[(size_t)(n0 + rowB) * Kd + kk + segB * 8]);
    };

    Acc acc[2][2], acc2[2][2];
#pragma unroll
    for (int i = 0; i < 2; i++)
#pragma unroll
        for (int j = 0; j < 2; j++)
#pragma unroll
            for (int r = 0; r < 8; r++) { acc[i][j].f[r] = 0.f; acc2[i][j].f[r] = 0.f; }

    int nsteps = Kd >> 5;
    stage(0, 0);
    wait_asynccnt0();
    __syncthreads();

    for (int step = 0; step < nsteps; step++) {
        int cur = step & 1;
        if (step + 1 < nsteps) stage(cur ^ 1, (step + 1) * 32);  // overlap DMA with WMMA

        // ---- fragments from LDS buffer `cur` ----
        const unsigned short* cA = &sA[cur][0];
        const unsigned short* cB = &sB[cur][0];
        FragB a[2], b[2], b2[2];
#pragma unroll
        for (int ms = 0; ms < 2; ms++) {
            int r = waveM * 32 + ms * 16 + lx;
            a[ms].u[0] = *(const uint4*)&cA[r * 32 + half * 16];
            a[ms].u[1] = *(const uint4*)&cA[r * 32 + half * 16 + 8];
        }
#pragma unroll
        for (int ns = 0; ns < 2; ns++) {
            int r = waveN * 32 + ns * 16 + lx;
            b[ns].u[0] = *(const uint4*)&cB[r * 32 + half * 16];
            b[ns].u[1] = *(const uint4*)&cB[r * 32 + half * 16 + 8];
            if constexpr (EPI == 3) {
                const unsigned short* cB2 = &sB2[cur][0];
                b2[ns].u[0] = *(const uint4*)&cB2[r * 32 + half * 16];
                b2[ns].u[1] = *(const uint4*)&cB2[r * 32 + half * 16 + 8];
            }
        }
#pragma unroll
        for (int ms = 0; ms < 2; ms++)
#pragma unroll
            for (int ns = 0; ns < 2; ns++) {
                acc[ms][ns].v = wmma_bf16(a[ms].v, b[ns].v, acc[ms][ns].v);
                if constexpr (EPI == 3)
                    acc2[ms][ns].v = wmma_bf16(a[ms].v, b2[ns].v, acc2[ms][ns].v);
            }
        wait_asynccnt0();     // next buffer's DMA landed
        __syncthreads();      // all waves done reading `cur`
    }

    // ---- epilogue ----
#pragma unroll
    for (int ms = 0; ms < 2; ms++)
#pragma unroll
        for (int ns = 0; ns < 2; ns++) {
            int col = n0 + waveN * 32 + ns * 16 + lx;
            float bv = bias ? bias[col] : 0.f;
#pragma unroll
            for (int r = 0; r < 8; r++) {
                int row = m0 + waveM * 32 + ms * 16 + r + half * 8;
                size_t o = (size_t)row * N + col;
                float v = acc[ms][ns].f[r] + bv;
                if constexpr (EPI == 0) {
                    Cf[o] = v;
                } else if constexpr (EPI == 1) {
                    Cf[o] = v + (res ? res[o] : 0.f);
                } else if constexpr (EPI == 2) {
                    Cb[o] = f2bf(v);
                } else {
                    float g = acc[ms][ns].f[r];
                    float val = acc2[ms][ns].f[r];
                    float sil = g / (1.f + __expf(-g));
                    Cb[o] = f2bf(sil * val);
                }
            }
        }
}

// ---------------------------------------------------------------------------
// RoPE in place on bf16 activations laid out (B*W, H*DH)
// ---------------------------------------------------------------------------
__global__ void k_rope(unsigned short* __restrict__ qk, const int* __restrict__ pos) {
    int idx = blockIdx.x * blockDim.x + threadIdx.x;  // B*W*H*32 threads
    if (idx >= BB * WW * HH * 32) return;
    int j = idx & 31;
    int h = (idx >> 5) & 7;
    int t = (idx >> 8) & 511;
    int b = idx >> 17;
    float p = (float)pos[t];
    float ang = p * __powf(10000.f, -((float)(2 * j)) / (float)DH);
    float c = __cosf(ang), s = __sinf(ang);
    size_t base = ((size_t)(b * WW + t)) * DD + h * DH + 2 * j;
    float x0 = bf2f(qk[base]), x1 = bf2f(qk[base + 1]);
    qk[base]     = f2bf(x0 * c - x1 * s);
    qk[base + 1] = f2bf(x0 * s + x1 * c);
}

// ---------------------------------------------------------------------------
// Flash-style causal attention, WMMA, per (b,h). Block = 4 waves, 64 q rows.
// K tile staged with async DMA; V tile transposed via LDS scalar stores.
// ---------------------------------------------------------------------------
__global__ __launch_bounds__(128)
void k_attn(const unsigned short* __restrict__ q,
            const unsigned short* __restrict__ k,
            const unsigned short* __restrict__ v,
            unsigned short* __restrict__ o) {
    __shared__ __align__(16) unsigned short kt[32 * 64];      // [s][dh]
    __shared__ __align__(16) unsigned short vt[64 * 32];      // [dh][s]
    __shared__ __align__(16) unsigned short pb[4][16 * 32];   // per-wave P

    int tid = threadIdx.x;
    int lane = tid & 31, wv = tid >> 5;
    int half = lane >> 4, lx = lane & 15;
    int bh = blockIdx.x;
    int b = bh / HH, hh = bh - b * HH;
    int q0 = blockIdx.y * 64;
    int qw = q0 + wv * 16;

    unsigned ktBase = lds_off(&kt[0]);

    // Q fragments for this wave (dh 0..63), kept in registers.
    FragB aq[2];
    {
        const unsigned short* qr = q + ((size_t)(b * WW + qw + lx)) * DD + hh * DH;
#pragma unroll
        for (int f = 0; f < 2; f++) {
            aq[f].u[0] = *(const uint4*)&qr[f * 32 + half * 16];
            aq[f].u[1] = *(const uint4*)&qr[f * 32 + half * 16 + 8];
        }
    }

    Acc oacc[4];
    float m_[8], l_[8];
#pragma unroll
    for (int nt = 0; nt < 4; nt++)
#pragma unroll
        for (int r = 0; r < 8; r++) oacc[nt].f[r] = 0.f;
#pragma unroll
    for (int r = 0; r < 8; r++) { m_[r] = -3e38f; l_[r] = 0.f; }

    for (int s0 = 0; s0 < q0 + 64; s0 += 32) {
        // stage K tile (32 x 64 bf16, row-major [s][dh]) via async DMA
#pragma unroll
        for (int i = 0; i < 2; i++) {
            int c = tid * 2 + i, s = c >> 3, seg = c & 7;
            async_copy_b128(ktBase + (unsigned)(s * 64 + seg * 8) * 2u,
                            &k[((size_t)(b * WW + s0 + s)) * DD + hh * DH + seg * 8]);
        }
        // stage transposed V tile: vt[dh][s]
        for (int i = 0; i < 16; i++) {
            int e = tid * 16 + i, s = e & 31, dh = e >> 5;
            vt[dh * 32 + s] = v[((size_t)(b * WW + s0 + s)) * DD + hh * DH + dh];
        }
        wait_asynccnt0();
        __syncthreads();

#pragma unroll
        for (int kk2 = 0; kk2 < 32; kk2 += 16) {
            // S = Q * K^T for 16 keys
            Acc S;
#pragma unroll
            for (int r = 0; r < 8; r++) S.f[r] = 0.f;
            FragB bk;
#pragma unroll
            for (int f = 0; f < 2; f++) {
                bk.u[0] = *(const uint4*)&kt[(kk2 + lx) * 64 + f * 32 + half * 16];
                bk.u[1] = *(const uint4*)&kt[(kk2 + lx) * 64 + f * 32 + half * 16 + 8];
                S.v = wmma_bf16(aq[f].v, bk.v, S.v);
            }
            // scale + causal mask; rows r+8*half, col = s0+kk2+lx
            float mt[8], ps[8];
#pragma unroll
            for (int r = 0; r < 8; r++) {
                float x = S.f[r] * 0.125f;
                int sg = s0 + kk2 + lx, qg = qw + r + half * 8;
                if (sg > qg) x = -1e30f;
                S.f[r] = x; mt[r] = x;
            }
#pragma unroll
            for (int off = 8; off > 0; off >>= 1)
#pragma unroll
                for (int r = 0; r < 8; r++)
                    mt[r] = fmaxf(mt[r], __shfl_xor(mt[r], off, 32));
            float rs[8];
#pragma unroll
            for (int r = 0; r < 8; r++) {
                float mn = fmaxf(m_[r], mt[r]);
                float al = __expf(m_[r] - mn);
                float p = __expf(S.f[r] - mn);
                ps[r] = p; rs[r] = p; m_[r] = mn;
                l_[r] *= al;
#pragma unroll
                for (int nt = 0; nt < 4; nt++) oacc[nt].f[r] *= al;
            }
#pragma unroll
            for (int off = 8; off > 0; off >>= 1)
#pragma unroll
                for (int r = 0; r < 8; r++) rs[r] += __shfl_xor(rs[r], off, 32);
#pragma unroll
            for (int r = 0; r < 8; r++) {
                l_[r] += rs[r];
                pb[wv][(r + half * 8) * 32 + kk2 + lx] = f2bf(ps[r]);
            }
        }
        // wave-local RAW on pb: drain DScnt before re-reading as fragments
        asm volatile("s_wait_dscnt 0x0" ::: "memory");

        // O += P(16x32) * V(32x64)
        FragB ap;
        ap.u[0] = *(const uint4*)&pb[wv][lx * 32 + half * 16];
        ap.u[1] = *(const uint4*)&pb[wv][lx * 32 + half * 16 + 8];
#pragma unroll
        for (int nt = 0; nt < 4; nt++) {
            FragB bv_;
            bv_.u[0] = *(const uint4*)&vt[(nt * 16 + lx) * 32 + half * 16];
            bv_.u[1] = *(const uint4*)&vt[(nt * 16 + lx) * 32 + half * 16 + 8];
            oacc[nt].v = wmma_bf16(ap.v, bv_.v, oacc[nt].v);
        }
        __syncthreads();
    }

    // write O (bf16)
#pragma unroll
    for (int nt = 0; nt < 4; nt++)
#pragma unroll
        for (int r = 0; r < 8; r++) {
            float ov = oacc[nt].f[r] / l_[r];
            o[((size_t)(b * WW + qw + r + half * 8)) * DD + hh * DH + nt * 16 + lx] = f2bf(ov);
        }
}

// ---------------------------------------------------------------------------
// q_win = mean_t h2[b,t,:]
// ---------------------------------------------------------------------------
__global__ __launch_bounds__(128)
void k_qwin(const float* __restrict__ h2, float* __restrict__ qwin) {
    int b = blockIdx.x;
    int d = blockIdx.y * 128 + threadIdx.x;
    float s = 0.f;
    for (int t = 0; t < WW; t++) s += h2[((size_t)(b * WW + t)) * DD + d];
    qwin[b * DD + d] = s * (1.f / (float)WW);
}

// ---------------------------------------------------------------------------
// Episodic memory + gate, one block per batch
// ---------------------------------------------------------------------------
__global__ __launch_bounds__(256)
void k_memcell(const float* __restrict__ qwin,
               const float* __restrict__ keys, const float* __restrict__ vals,
               const float* __restrict__ age, const float* __restrict__ strength,
               const float* __restrict__ rms_read,
               const float* __restrict__ Wwk, const float* __restrict__ bwk,
               const float* __restrict__ Wwv, const float* __restrict__ bwv,
               const float* __restrict__ Wws, const float* __restrict__ bws,
               const float* __restrict__ Wg1, const float* __restrict__ bg1,
               const float* __restrict__ Wg2, const float* __restrict__ bg2,
               float* __restrict__ out_keys, float* __restrict__ out_vals,
               float* __restrict__ out_age, float* __restrict__ out_sn,
               float* __restrict__ out_gate, float* __restrict__ gr) {
    int b = blockIdx.x, tid = threadIdx.x;
    __shared__ float sq[512], swk[512], swv[512], sread[512], shid[512];
    __shared__ float slog[64], ssimw[64], swread[64];
    __shared__ float red[256];
    __shared__ float sc[8];

    for (int d = tid; d < 512; d += 256) sq[d] = qwin[b * 512 + d];
    __syncthreads();

    // 1/(||q||+eps)
    {
        float ss = 0.f;
        for (int d = tid; d < 512; d += 256) ss += sq[d] * sq[d];
        red[tid] = ss; __syncthreads();
        for (int s = 128; s > 0; s >>= 1) { if (tid < s) red[tid] += red[tid + s]; __syncthreads(); }
        if (tid == 0) sc[0] = 1.f / (sqrtf(red[0]) + 1e-6f);
        __syncthreads();
    }
    // write_key / write_val matvecs + ws dot
    for (int o = tid; o < 512; o += 256) {
        float aK = bwk[o], aV = bwv[o];
        for (int d = 0; d < 512; d++) { aK += sq[d] * Wwk[d * 512 + o]; aV += sq[d] * Wwv[d * 512 + o]; }
        swk[o] = aK; swv[o] = aV;
    }
    {
        float ps = 0.f;
        for (int d = tid; d < 512; d += 256) ps += sq[d] * Wws[d];
        red[tid] = ps; __syncthreads();
        for (int s = 128; s > 0; s >>= 1) { if (tid < s) red[tid] += red[tid + s]; __syncthreads(); }
        if (tid == 0) sc[1] = 1.f / (1.f + __expf(-(red[0] + bws[0])));  // ws in [0,1]
        __syncthreads();
    }
    // ||write_key||
    {
        float ss = 0.f;
        for (int d = tid; d < 512; d += 256) ss += swk[d] * swk[d];
        red[tid] = ss; __syncthreads();
        for (int s = 128; s > 0; s >>= 1) { if (tid < s) red[tid] += red[tid + s]; __syncthreads(); }
        if (tid == 0) sc[2] = 1.f / (sqrtf(red[0]) + 1e-6f);
        __syncthreads();
    }
    // per-slot dots (4 threads per slot)
    int slot = tid >> 2, g = tid & 3;
    {
        float dq = 0.f, dw = 0.f, kk = 0.f;
        const float* kp = keys + ((size_t)b * KK + slot) * 512;
        for (int d = g; d < 512; d += 4) { float kvv = kp[d]; dq += kvv * sq[d]; dw += kvv * swk[d]; kk += kvv * kvv; }
        for (int off = 1; off < 4; off <<= 1) {
            dq += __shfl_xor(dq, off, 32); dw += __shfl_xor(dw, off, 32); kk += __shfl_xor(kk, off, 32);
        }
        if (g == 0) {
            float invk = 1.f / (sqrtf(kk) + 1e-6f);
            float simr = dq * invk * sc[0];
            float simw = dw * invk * sc[2];
            float st = strength[b * KK + slot];
            float lg = simr + 0.5f * logf(fminf(fmaxf(st, 1e-3f), 1e9f))
                     - 0.02f * age[b * KK + slot] + ((st > 1e-3f) ? 0.f : -1000.f);
            slog[slot] = lg; ssimw[slot] = simw;
        }
    }
    __syncthreads();
    // softmax over logits + argmax of sim_w (serial, 64 elems)
    if (tid == 0) {
        float mx = -3e38f;
        for (int i = 0; i < KK; i++) mx = fmaxf(mx, slog[i]);
        float sum = 0.f;
        for (int i = 0; i < KK; i++) { float e = __expf(slog[i] - mx); swread[i] = e; sum += e; }
        float inv = 1.f / sum;
        for (int i = 0; i < KK; i++) swread[i] *= inv;
        int am = 0; float bs = ssimw[0];
        for (int i = 1; i < KK; i++) if (ssimw[i] > bs) { bs = ssimw[i]; am = i; }
        sc[3] = (float)am; sc[4] = bs;
    }
    __syncthreads();
    int widx = (int)sc[3];
    float wsv = sc[1];
    // read vector
    for (int d = tid; d < 512; d += 256) {
        float r = 0.f;
        for (int s = 0; s < KK; s++) r += swread[s] * vals[((size_t)b * KK + s) * 512 + d];
        sread[d] = r;
    }
    __syncthreads();
    {
        float ss = 0.f;
        for (int d = tid; d < 512; d += 256) ss += sread[d] * sread[d];
        red[tid] = ss; __syncthreads();
        for (int s = 128; s > 0; s >>= 1) { if (tid < s) red[tid] += red[tid + s]; __syncthreads(); }
        if (tid == 0) sc[5] = rsqrtf(red[0] / 512.f + 1e-6f);
        __syncthreads();
        for (int d = tid; d < 512; d += 256) sread[d] = sread[d] * sc[5] * rms_read[d];
        __syncthreads();
    }
    // memory updates (write_w is the hard one-hot: stop_grad(hard-soft)+soft == hard)
    {
        float re = (slot == widx) ? wsv * 0.5f : 0.f;
        const float* kp = keys + ((size_t)b * KK + slot) * 512;
        const float* vp = vals + ((size_t)b * KK + slot) * 512;
        float cn = 0.f;
        for (int d = g; d < 512; d += 4) { float c = (1.f - re) * kp[d] + re * swk[d]; cn += c * c; }
        for (int off = 1; off < 4; off <<= 1) cn += __shfl_xor(cn, off, 32);
        float invc = 1.f / (sqrtf(cn) + 1e-6f);
        for (int d = g; d < 512; d += 4) {
            float c = (1.f - re) * kp[d] + re * swk[d];
            out_keys[((size_t)b * KK + slot) * 512 + d] = c * invc;
            out_vals[((size_t)b * KK + slot) * 512 + d] = (1.f - re) * vp[d] + re * swv[d];
        }
        if (g == 0) {
            float hard = (slot == widx) ? 1.f : 0.f;
            out_age[b * KK + slot] = (age[b * KK + slot] + 1.f) * (1.f - hard);
            float sn0 = strength[b * KK + slot] * 0.995f;
            float snv = sn0 + hard * wsv * (1.f - sn0);
            out_sn[b * KK + slot] = fminf(fmaxf(snv, 1e-3f), 1.f);
        }
    }
    __syncthreads();
    // gate MLP: xg = [q_win, ws, novelty]
    float novelty = 1.f - sc[4];
    for (int o = tid; o < 512; o += 256) {
        float a = bg1[o];
        for (int i = 0; i < 512; i++) a += sq[i] * Wg1[i * 512 + o];
        a += wsv * Wg1[512 * 512 + o] + novelty * Wg1[513 * 512 + o];
        shid[o] = a / (1.f + __expf(-a));  // silu
    }
    __syncthreads();
    {
        float ps = 0.f;
        for (int d = tid; d < 512; d += 256) ps += shid[d] * Wg2[d];
        red[tid] = ps; __syncthreads();
        for (int s = 128; s > 0; s >>= 1) { if (tid < s) red[tid] += red[tid + s]; __syncthreads(); }
        if (tid == 0) { float gt = 1.f / (1.f + __expf(-(red[0] + bg2[0]))); sc[6] = gt; out_gate[b] = gt; }
        __syncthreads();
    }
    for (int d = tid; d < 512; d += 256) gr[b * 512 + d] = sc[6] * sread[d];
}

// ---------------------------------------------------------------------------
// out = h2 + gr[b] broadcast over t
// ---------------------------------------------------------------------------
__global__ void k_final_add(const float* __restrict__ h2, const float* __restrict__ gr,
                            float* __restrict__ out) {
    int idx = blockIdx.x * blockDim.x + threadIdx.x;
    if (idx >= MM * DD) return;
    int b = idx >> 18;          // / (512*512)
    int d = idx & 511;
    out[idx] = h2[idx] + gr[b * 512 + d];
}

// ---------------------------------------------------------------------------
// Host: kernel_launch
// ---------------------------------------------------------------------------
extern "C" void kernel_launch(void* const* d_in, const int* in_sizes, int n_in,
                              void* d_out, int out_size, void* d_ws, size_t ws_size,
                              hipStream_t stream) {
    const float* x      = (const float*)d_in[0];
    const float* keys   = (const float*)d_in[1];
    const float* vals   = (const float*)d_in[2];
    const float* age    = (const float*)d_in[3];
    const float* streng = (const float*)d_in[4];
    const int*   pos    = (const int*)d_in[5];
    const float* rms1   = (const float*)d_in[6];
    const float* rms_kv = (const float*)d_in[7];
    const float* rms2   = (const float*)d_in[8];
    const float* rms_rd = (const float*)d_in[9];
    const float* Wq = (const float*)d_in[10]; const float* bq = (const float*)d_in[11];
    const float* Wk = (const float*)d_in[12]; const float* bk = (const float*)d_in[13];
    const float* Wv = (const float*)d_in[14]; const float* bv = (const float*)d_in[15];
    const float* Wo = (const float*)d_in[16]; const float* bo = (const float*)d_in[17];
    const float* W1 = (const float*)d_in[18];
    const float* W2 = (const float*)d_in[19];
    const float* Wwk = (const float*)d_in[20]; const float* bwk = (const float*)d_in[21];
    const float* Wwv = (const float*)d_in[22]; const float* bwv = (const float*)d_in[23];
    const float* Wws = (const float*)d_in[24]; const float* bws = (const float*)d_in[25];
    const float* Wg1 = (const float*)d_in[26]; const float* bg1 = (const float*)d_in[27];
    const float* Wg2 = (const float*)d_in[28]; const float* bg2 = (const float*)d_in[29];

    // workspace carve-out
    char* ws = (char*)d_ws;
    size_t off = 0;
    auto carve = [&](size_t bytes) -> char* {
        char* p = ws + off;
        off = (off + bytes + 255) & ~(size_t)255;
        return p;
    };
    unsigned short* wtq  = (unsigned short*)carve((size_t)DD * DD * 2);
    unsigned short* wtk  = (unsigned short*)carve((size_t)DD * DD * 2);
    unsigned short* wtv  = (unsigned short*)carve((size_t)DD * DD * 2);
    unsigned short* wto  = (unsigned short*)carve((size_t)DD * DD * 2);
    unsigned short* w1t  = (unsigned short*)carve((size_t)4096 * DD * 2);
    unsigned short* w2t  = (unsigned short*)carve((size_t)DD * 2048 * 2);
    unsigned short* nx   = (unsigned short*)carve((size_t)MM * DD * 2);
    unsigned short* nkv  = (unsigned short*)carve((size_t)MM * DD * 2);
    unsigned short* qb   = (unsigned short*)carve((size_t)MM * DD * 2);
    unsigned short* kb   = (unsigned short*)carve((size_t)MM * DD * 2);
    unsigned short* vb   = (unsigned short*)carve((size_t)MM * DD * 2);
    unsigned short* ob   = (unsigned short*)carve((size_t)MM * DD * 2);
    float*          h1   = (float*)carve((size_t)MM * DD * 4);
    unsigned short* n2   = (unsigned short*)carve((size_t)MM * DD * 2);
    unsigned short* hact = (unsigned short*)carve((size_t)MM * 2048 * 2);
    float*          h2   = (float*)carve((size_t)MM * DD * 4);
    float*          qwin = (float*)carve((size_t)BB * DD * 4);
    float*          gr   = (float*)carve((size_t)BB * DD * 4);
    (void)ws_size; (void)n_in; (void)in_sizes; (void)out_size;

    float* out_main = (float*)d_out;
    float* out_keys = out_main + (size_t)BB * WW * DD;
    float* out_vals = out_keys + (size_t)BB * KK * DD;
    float* out_age  = out_vals + (size_t)BB * KK * DD;
    float* out_sn   = out_age + (size_t)BB * KK;
    float* out_gate = out_sn + (size_t)BB * KK;

    // 1) weight prep (transpose + bf16)
    k_transpose_bf16<<<(DD * DD + 255) / 256, 256, 0, stream>>>(Wq, wtq, DD, DD);
    k_transpose_bf16<<<(DD * DD + 255) / 256, 256, 0, stream>>>(Wk, wtk, DD, DD);
    k_transpose_bf16<<<(DD * DD + 255) / 256, 256, 0, stream>>>(Wv, wtv, DD, DD);
    k_transpose_bf16<<<(DD * DD + 255) / 256, 256, 0, stream>>>(Wo, wto, DD, DD);
    k_transpose_bf16<<<(DD * 4096 + 255) / 256, 256, 0, stream>>>(W1, w1t, DD, 4096);
    k_transpose_bf16<<<(2048 * DD + 255) / 256, 256, 0, stream>>>(W2, w2t, 2048, DD);

    // 2) RMS norms of x
    k_rmsnorm_bf16<<<MM, 256, 0, stream>>>(x, rms1, nx);
    k_rmsnorm_bf16<<<MM, 256, 0, stream>>>(x, rms_kv, nkv);

    // 3) QKV projections (bf16 out, +bias)
    dim3 g512(DD / 64, MM / 128);
    k_gemm<2><<<g512, 256, 0, stream>>>(nx,  wtq, nullptr, bq, nullptr, nullptr, qb, MM, DD, DD);
    k_gemm<2><<<g512, 256, 0, stream>>>(nkv, wtk, nullptr, bk, nullptr, nullptr, kb, MM, DD, DD);
    k_gemm<2><<<g512, 256, 0, stream>>>(nkv, wtv, nullptr, bv, nullptr, nullptr, vb, MM, DD, DD);

    // 4) RoPE on q, k
    k_rope<<<(BB * WW * HH * 32 + 255) / 256, 256, 0, stream>>>(qb, pos);
    k_rope<<<(BB * WW * HH * 32 + 255) / 256, 256, 0, stream>>>(kb, pos);

    // 5) attention
    k_attn<<<dim3(BB * HH, WW / 64), 128, 0, stream>>>(qb, kb, vb, ob);

    // 6) h1 = x + o @ Wo + bo
    k_gemm<1><<<g512, 256, 0, stream>>>(ob, wto, nullptr, bo, x, h1, nullptr, MM, DD, DD);

    // 7) n2 = rms(h1)
    k_rmsnorm_bf16<<<MM, 256, 0, stream>>>(h1, rms2, n2);

    // 8) hact = silu(n2@W1_g) * (n2@W1_v)   (fused dual GEMM, bf16 out)
    dim3 g2048(2048 / 64, MM / 128);
    k_gemm<3><<<g2048, 256, 0, stream>>>(n2, w1t, w1t + (size_t)2048 * DD, nullptr,
                                         nullptr, nullptr, hact, MM, 2048, DD);

    // 9) h2 = h1 + hact @ W2
    k_gemm<1><<<g512, 256, 0, stream>>>(hact, w2t, nullptr, nullptr, h1, h2, nullptr, MM, DD, 2048);

    // 10) q_win mean
    k_qwin<<<dim3(BB, DD / 128), 128, 0, stream>>>(h2, qwin);

    // 11) episodic memory + gate
    k_memcell<<<BB, 256, 0, stream>>>(qwin, keys, vals, age, streng, rms_rd,
                                      Wwk, bwk, Wwv, bwv, Wws, bws, Wg1, bg1, Wg2, bg2,
                                      out_keys, out_vals, out_age, out_sn, out_gate, gr);

    // 12) out = h2 + gate*read
    k_final_add<<<(MM * DD + 255) / 256, 256, 0, stream>>>(h2, gr, out_main);
}
